// ConvCharEmbedding_35442070126625
// MI455X (gfx1250) — matrix operand order repaired and checked by reading, not compiled, not run on hardware
//
#include <hip/hip_runtime.h>

typedef _Float16 v16h __attribute__((ext_vector_type(16)));
typedef _Float16 v8h  __attribute__((ext_vector_type(8)));
typedef _Float16 v4h  __attribute__((ext_vector_type(4)));
typedef float    v8f  __attribute__((ext_vector_type(8)));

#define NWAVES   8
#define BLOCK    (NWAVES * 32)
#define XROW     136                 // padded xpad row stride in halves (272 B -> conflict-free rows)
#define XROWS    20                  // chars -2..17 (rows 0,1,18,19 are zero pad)
#define XWORD    (XROWS * XROW)      // halves per staged word
#define WELEMS   (5 * 8 * 4 * 32 * 16)   // k * otile * kchunk * lane * halves = 81920 halves (160 KB)

// One wave computes TWO words' Y[16 chars, 128 out-ch] per iteration so each
// B-fragment ds_load feeds two v_wmma_f32_16x16x32_f16 (halves LDS traffic).
// Per pair: 5 (tap) x 4 (K=32 chunk) x 8 (N-tile) x 2 (word) WMMAs, then
// max-pool over chars + bias.
__global__ __launch_bounds__(BLOCK)
void conv_char_emb_kernel(const int*  __restrict__ ids,     // [N,16] char ids
                          const float* __restrict__ emb,    // [262,128]
                          const float* __restrict__ convw,  // [128 o][128 i][5 k]
                          const float* __restrict__ convb,  // [128]
                          float* __restrict__ out,          // [N,128]
                          int nWords)
{
    __shared__ _Float16 wlds[WELEMS];                     // B fragments, pre-swizzled
    __shared__ _Float16 xpad[NWAVES * 2 * XWORD];         // per-wave, 2 words of padded f16 embeddings

    const int tid  = threadIdx.x;
    const int lane = tid & 31;
    const int wave = tid >> 5;
    const int hi   = lane >> 4;   // which 16-lane half
    const int lm   = lane & 15;

    // ---- Stage conv weights into LDS in exact WMMA B-fragment order (f32 -> f16).
    // B fragment (k, otile, chunk): lane L holds halves h=0..15 with
    //   N = L&15 (= o_local), K = 16*(L>>4) + h (= i_local), contiguous per lane.
    for (int idx = tid; idx < WELEMS; idx += BLOCK) {
        int h     =  idx        & 15;
        int ln    = (idx >> 4)  & 31;
        int chunk = (idx >> 9)  & 3;
        int ot    = (idx >> 11) & 7;
        int k     =  idx >> 14;
        int o = ot * 16 + (ln & 15);
        int i = chunk * 32 + (ln >> 4) * 16 + h;
        wlds[idx] = (_Float16)convw[(o * 128 + i) * 5 + k];
    }
    __syncthreads();

    _Float16* xp0 = xpad + wave * (2 * XWORD);
    _Float16* xp1 = xp0 + XWORD;
    const int pairBase   = (blockIdx.x * NWAVES + wave) * 2;
    const int pairStride = gridDim.x * NWAVES * 2;

    #pragma clang loop unroll(disable)
    for (int n = pairBase; n + 1 < nWords; n += pairStride) {
        // ---- Stage padded, f16-converted embeddings for words n and n+1.
        // xpad row r = embedding of char (r-2); rows 0,1,18,19 are zeros.
        for (int r = 0; r < XROWS; ++r) {
            v4h hv0 = {}, hv1 = {};
            if (r >= 2 && r < 18) {
                int id0 = ids[n * 16 + (r - 2)];
                int id1 = ids[(n + 1) * 16 + (r - 2)];
                const float4 f0 = *(const float4*)(emb + (long)id0 * 128 + lane * 4);
                const float4 f1 = *(const float4*)(emb + (long)id1 * 128 + lane * 4);
                hv0[0] = (_Float16)f0.x; hv0[1] = (_Float16)f0.y;
                hv0[2] = (_Float16)f0.z; hv0[3] = (_Float16)f0.w;
                hv1[0] = (_Float16)f1.x; hv1[1] = (_Float16)f1.y;
                hv1[2] = (_Float16)f1.z; hv1[3] = (_Float16)f1.w;
            }
            *(v4h*)(xp0 + r * XROW + lane * 4) = hv0;
            *(v4h*)(xp1 + r * XROW + lane * 4) = hv1;
        }
        // xpad is private to this wave; DS ops are in-order within a wave.

        // Opaque zero offsets: keep the B (weight) ds_loads inside this loop
        // (LICM across words would materialize all fragments and spill).
        unsigned bzero = 0, azero = 0;
        asm volatile("" : "+v"(bzero), "+v"(azero));
        const _Float16* wb  = wlds + bzero;
        const _Float16* xb0 = xp0 + azero;
        const _Float16* xb1 = xp1 + azero;

        v8f acc0[8] = {};   // word n   : 8 N-tiles of C/D (16x16 f32)
        v8f acc1[8] = {};   // word n+1 : 8 N-tiles of C/D (16x16 f32)

        #pragma unroll
        for (int k = 0; k < 5; ++k) {
            #pragma unroll
            for (int chunk = 0; chunk < 4; ++chunk) {
                // A fragment (16x32 f16): M = char c = lane&15 (row c+k of xpad),
                // lanes 0-15: a[0..15]=K0..15; lanes 16-31: a[0..15]=K16..31.
                const int aoff = (lm + k) * XROW + chunk * 32 + hi * 16;
                v8h a00 = *(const v8h*)(xb0 + aoff);
                v8h a01 = *(const v8h*)(xb0 + aoff + 8);
                v8h a10 = *(const v8h*)(xb1 + aoff);
                v8h a11 = *(const v8h*)(xb1 + aoff + 8);
                v16h a0 = __builtin_shufflevector(a00, a01,
                           0,1,2,3,4,5,6,7,8,9,10,11,12,13,14,15);
                v16h a1 = __builtin_shufflevector(a10, a11,
                           0,1,2,3,4,5,6,7,8,9,10,11,12,13,14,15);
                #pragma unroll
                for (int ot = 0; ot < 8; ++ot) {
                    const _Float16* bp =
                        wb + (((k * 8 + ot) * 4 + chunk) * 32 + lane) * 16;
                    v8h b0 = *(const v8h*)bp;
                    v8h b1 = *(const v8h*)(bp + 8);
                    v16h b = __builtin_shufflevector(b0, b1,
                              0,1,2,3,4,5,6,7,8,9,10,11,12,13,14,15);
                    acc0[ot] = __builtin_amdgcn_wmma_f32_16x16x32_f16(
                        false, a0, false, b, (short)0, acc0[ot], false, false);
                    acc1[ot] = __builtin_amdgcn_wmma_f32_16x16x32_f16(
                        false, a1, false, b, (short)0, acc1[ot], false, false);
                }
            }
        }

        // ---- Max over chars (M), add bias, store both words.
        // D layout: lane -> N = lane&15; VGPR v -> M = v + 8*(lane>>4).
        #pragma unroll
        for (int ot = 0; ot < 8; ++ot) {
            float m0 = acc0[ot][0];
            float m1 = acc1[ot][0];
            #pragma unroll
            for (int v = 1; v < 8; ++v) {
                m0 = fmaxf(m0, acc0[ot][v]);
                m1 = fmaxf(m1, acc1[ot][v]);
            }
            m0 = fmaxf(m0, __shfl_xor(m0, 16, 32));   // combine char halves 0-7 / 8-15
            m1 = fmaxf(m1, __shfl_xor(m1, 16, 32));
            if (lane < 16) {
                int o = ot * 16 + lane;
                float b = convb[o];
                out[(long)n * 128 + o]       = m0 + b;
                out[(long)(n + 1) * 128 + o] = m1 + b;
            }
        }
    }
}

extern "C" void kernel_launch(void* const* d_in, const int* in_sizes, int n_in,
                              void* d_out, int out_size, void* d_ws, size_t ws_size,
                              hipStream_t stream) {
    const int*   ids   = (const int*)d_in[0];     // input [B,W,C]
    // d_in[1] = lengths: unused by the reference
    const float* emb   = (const float*)d_in[2];   // [262,128]
    const float* convw = (const float*)d_in[3];   // [128,128,5]
    const float* convb = (const float*)d_in[4];   // [128]
    float* out = (float*)d_out;

    const int nWords = in_sizes[0] / 16;          // B*W = 32768 (even)

    conv_char_emb_kernel<<<256, BLOCK, 0, stream>>>(ids, emb, convw, convb, out, nWords);
}
